// VGAE_31018253811968
// MI455X (gfx1250) — matrix operand order repaired and checked by reading, not compiled, not run on hardware
//
#include <hip/hip_runtime.h>
#include <hip/hip_bf16.h>
#include <cstdint>

#define NODES 12288
#define IN_F  256
#define HID   128
#define DZ    64

typedef __attribute__((ext_vector_type(16))) _Float16 v16h;
typedef __attribute__((ext_vector_type(8)))  _Float16 v8h;
typedef __attribute__((ext_vector_type(8)))  float    v8f;

// Only the device pass has the AMDGCN WMMA builtins; don't probe on host.
#if defined(__HIP_DEVICE_COMPILE__) && !__has_builtin(__builtin_amdgcn_wmma_f32_16x16x32_f16)
#error "__builtin_amdgcn_wmma_f32_16x16x32_f16 not available on this toolchain (device)"
#endif

// ---------------------------------------------------------------------------
// Degree count: deg[dst[e]] += 1
// ---------------------------------------------------------------------------
__global__ void deg_kernel(const int* __restrict__ dst, float* __restrict__ deg, int E) {
    int i = blockIdx.x * blockDim.x + threadIdx.x;
    int stride = gridDim.x * blockDim.x;
    for (; i < E; i += stride) {
        atomicAdd(&deg[dst[i]], 1.0f);
    }
}

// norm = 1/sqrt(max(deg,1)), in place
__global__ void norm_kernel(float* __restrict__ d, int n) {
    int i = blockIdx.x * blockDim.x + threadIdx.x;
    if (i < n) d[i] = rsqrtf(fmaxf(d[i], 1.0f));
}

// Split fp32 -> hi/lo f16 pair: v = hi + lo with hi = f16(v), lo = f16(v - hi)
__global__ void split_f16(const float* __restrict__ src, _Float16* __restrict__ hi,
                          _Float16* __restrict__ lo, int n) {
    int i = blockIdx.x * blockDim.x + threadIdx.x;
    if (i >= n) return;
    float v = src[i];
    _Float16 h = (_Float16)v;
    hi[i] = h;
    lo[i] = (_Float16)(v - (float)h);
}

// ---------------------------------------------------------------------------
// Split-f16 WMMA GEMM: C[M,Nn] = A[M,K] @ B  (fp32-class accuracy via
// Ah@Bh + Ah@Bl + Al@Bh, all accumulated in one f32 WMMA accumulator).
//   TRANSB=false: B is row-major [K,Nn]
//   TRANSB=true : B is row-major [Nn,K]  (compute A @ B^T)
//   SCALE      : multiply output row r by rowscale[r] (branchless, templated)
//   NT         : non-temporal output stores (streaming, don't pollute L2)
// One wave -> one 16x16 C tile; K multiple of 32; M,Nn multiples of 16.
// Fragment addressing (contiguous k order used here for both A and B, 16-bit 16x32 A / 32x16 B, wave32):
//   A elems 0..7  = A[m, k0 + 16*half + 0..7]     (16B contiguous)
//   A elems 8..15 = A[m, k0 + 16*half + 8..15]    (16B contiguous)
//   B elem j      = B[k0 + 16*half + j, col]      (32B contiguous when TRANSB)
// ---------------------------------------------------------------------------
template<bool TRANSB, bool SCALE, bool NT>
__global__ void gemm_wmma_split(const _Float16* __restrict__ Ah, const _Float16* __restrict__ Al,
                                const _Float16* __restrict__ Bh, const _Float16* __restrict__ Bl,
                                float* __restrict__ C, int M, int Nn, int K,
                                const float* __restrict__ rowscale) {
    (void)M;
    const int lane = threadIdx.x;                       // 0..31
    const int wave = threadIdx.y;                       // 0..7
    const int colTile = blockIdx.x * 8 + wave;
    const int rowTile = blockIdx.y;
    const int nColTiles = Nn >> 4;
    if (colTile >= nColTiles) return;                   // whole-wave uniform exit

    const int half = lane >> 4;                         // 0 or 1
    const int l16  = lane & 15;
    const int arow = (rowTile << 4) + l16;              // A row this lane reads
    const int col  = (colTile << 4) + l16;              // output column this lane owns

    v8f c = {};
    for (int k0 = 0; k0 < K; k0 += 32) {
        // ---- A fragments (hi, lo): two contiguous b128 loads each ----
        const size_t abase = (size_t)arow * K + k0 + (half << 4);
        v8h ah0 = *(const v8h*)(Ah + abase);
        v8h ah1 = *(const v8h*)(Ah + abase + 8);
        v8h al0 = *(const v8h*)(Al + abase);
        v8h al1 = *(const v8h*)(Al + abase + 8);
        v16h ah = __builtin_shufflevector(ah0, ah1, 0,1,2,3,4,5,6,7,8,9,10,11,12,13,14,15);
        v16h al = __builtin_shufflevector(al0, al1, 0,1,2,3,4,5,6,7,8,9,10,11,12,13,14,15);
        // ---- B fragments (hi, lo) ----
        v16h bh, bl;
        if (TRANSB) {
            const size_t bbase = (size_t)col * K + k0 + (half << 4);   // 32B contiguous
            v8h bh0 = *(const v8h*)(Bh + bbase);
            v8h bh1 = *(const v8h*)(Bh + bbase + 8);
            v8h bl0 = *(const v8h*)(Bl + bbase);
            v8h bl1 = *(const v8h*)(Bl + bbase + 8);
            bh = __builtin_shufflevector(bh0, bh1, 0,1,2,3,4,5,6,7,8,9,10,11,12,13,14,15);
            bl = __builtin_shufflevector(bl0, bl1, 0,1,2,3,4,5,6,7,8,9,10,11,12,13,14,15);
        } else {
            const int kb = k0 + (half << 4);
            #pragma unroll
            for (int j = 0; j < 16; ++j) {
                bh[j] = Bh[(size_t)(kb + j) * Nn + col];
                bl[j] = Bl[(size_t)(kb + j) * Nn + col];
            }
        }
        // ---- three-product split accumulation ----
        c = __builtin_amdgcn_wmma_f32_16x16x32_f16(false, ah, false, bh, (short)0, c, false, false);
        c = __builtin_amdgcn_wmma_f32_16x16x32_f16(false, ah, false, bl, (short)0, c, false, false);
        c = __builtin_amdgcn_wmma_f32_16x16x32_f16(false, al, false, bh, (short)0, c, false, false);
    }

    // ---- branchless epilogue: c[r] -> row (rowTile*16 + r + 8*half), column col ----
    float sc[8];
    if (SCALE) {
        #pragma unroll
        for (int r = 0; r < 8; ++r)
            sc[r] = rowscale[(rowTile << 4) + r + (half << 3)];
    }
    #pragma unroll
    for (int r = 0; r < 8; ++r) {
        const int orow = (rowTile << 4) + r + (half << 3);
        float v = c[r];
        if (SCALE) v *= sc[r];
        float* p = C + (size_t)orow * Nn + col;
        if (NT) __builtin_nontemporal_store(v, p);
        else    *p = v;
    }
}

// ---------------------------------------------------------------------------
// Edge scatter-sum: agg[dst[e], f] += t[src[e], f]   (F = 1<<fshift)
// ---------------------------------------------------------------------------
__global__ void scatter_add(const float* __restrict__ t, float* __restrict__ agg,
                            const int* __restrict__ src, const int* __restrict__ dst,
                            long long total, int F, int fshift) {
    long long idx = (long long)blockIdx.x * blockDim.x + threadIdx.x;
    if (idx >= total) return;
    const int e = (int)(idx >> fshift);
    const int f = (int)(idx & (F - 1));
    atomicAdd(&agg[(size_t)dst[e] * F + f], t[(size_t)src[e] * F + f]);
}

// a[i] *= norm[i >> fshift]   (post-aggregation norm, in place)
__global__ void rowscale_kernel(float* __restrict__ a, const float* __restrict__ norm,
                                int n, int fshift) {
    int i = blockIdx.x * blockDim.x + threadIdx.x;
    if (i < n) a[i] *= norm[i >> fshift];
}

// mean = relu(aggm*norm); logstd = relu(aggs*norm); Z = noise*exp(logstd)+mean
__global__ void z_kernel(const float* __restrict__ aggm, const float* __restrict__ aggs,
                         const float* __restrict__ norm, const float* __restrict__ noise,
                         float* __restrict__ Z, int n) {
    int i = blockIdx.x * blockDim.x + threadIdx.x;
    if (i >= n) return;
    const float nr   = norm[i >> 6];                    // DZ == 64
    const float mean = fmaxf(aggm[i] * nr, 0.0f);
    const float ls   = fmaxf(aggs[i] * nr, 0.0f);
    Z[i] = noise[i] * __expf(ls) + mean;
}

// ---------------------------------------------------------------------------
extern "C" void kernel_launch(void* const* d_in, const int* in_sizes, int n_in,
                              void* d_out, int out_size, void* d_ws, size_t ws_size,
                              hipStream_t stream) {
    (void)n_in; (void)out_size; (void)ws_size;
    const float* X     = (const float*)d_in[0];   // [N, 256]
    const float* W0    = (const float*)d_in[1];   // [256, 128]
    const float* Wm    = (const float*)d_in[2];   // [128, 64]
    const float* Wsd   = (const float*)d_in[3];   // [128, 64]
    const float* noise = (const float*)d_in[4];   // [N, 64]
    const int*   edges = (const int*)d_in[5];     // [2, E] flattened
    const int E = in_sizes[5] / 2;
    const int* src = edges;
    const int* dst = edges + E;
    float* out = (float*)d_out;                   // [N, N]

    // ---- workspace layout: fp32 region first (zeroed part at the very front) ----
    float* ws   = (float*)d_ws;
    float* norm = ws;                               // N        (deg, then norm)
    float* agg1 = norm + NODES;                     // N*HID    (zeroed)
    float* agg2 = agg1 + (size_t)NODES * HID;       // N*HID    (zeroed; m | s halves)
    float* t1   = agg2 + (size_t)NODES * HID;       // N*HID
    float* t2   = t1   + (size_t)NODES * HID;       // N*HID    (meanpre | logstdpre)
    float* Zf   = t2   + (size_t)NODES * HID;       // N*DZ
    // ---- f16 hi/lo buffers after the fp32 region (region size N*577 floats, 256B aligned) ----
    _Float16* hbase = (_Float16*)(Zf + (size_t)NODES * DZ);
    _Float16* Xh  = hbase;                          // N*IN_F
    _Float16* Xl  = Xh  + (size_t)NODES * IN_F;
    _Float16* W0h = Xl  + (size_t)NODES * IN_F;     // IN_F*HID
    _Float16* W0l = W0h + (size_t)IN_F * HID;
    _Float16* Hh  = W0l + (size_t)IN_F * HID;       // N*HID
    _Float16* Hl  = Hh  + (size_t)NODES * HID;
    _Float16* Wmh = Hl  + (size_t)NODES * HID;      // HID*DZ
    _Float16* Wml = Wmh + (size_t)HID * DZ;
    _Float16* Wsh = Wml + (size_t)HID * DZ;
    _Float16* Wsl = Wsh + (size_t)HID * DZ;
    _Float16* Zh  = Wsl + (size_t)HID * DZ;         // N*DZ
    _Float16* Zl  = Zh  + (size_t)NODES * DZ;

    // zero deg + agg1 + agg2 (graph-capturable memset node)
    (void)hipMemsetAsync(ws, 0, sizeof(float) * (size_t)NODES * (1 + 2 * HID), stream);

    // degree -> norm
    deg_kernel<<<1024, 256, 0, stream>>>(dst, norm, E);
    norm_kernel<<<(NODES + 255) / 256, 256, 0, stream>>>(norm, NODES);

    const dim3 blk(32, 8);   // 8 waves / block, one 16x16 tile each

    // ---- layer 1: t1 = (X @ W0) * norm ; agg1 = scatter(t1) ; hidden = agg1*norm ----
    split_f16<<<(NODES * IN_F + 255) / 256, 256, 0, stream>>>(X, Xh, Xl, NODES * IN_F);
    split_f16<<<(IN_F * HID + 255) / 256, 256, 0, stream>>>(W0, W0h, W0l, IN_F * HID);
    gemm_wmma_split<false, true, false><<<dim3(1, NODES / 16), blk, 0, stream>>>(
        Xh, Xl, W0h, W0l, t1, NODES, HID, IN_F, norm);
    {
        long long tot = (long long)E * HID;
        scatter_add<<<(unsigned)((tot + 255) / 256), 256, 0, stream>>>(
            t1, agg1, src, dst, tot, HID, 7);
    }
    rowscale_kernel<<<(NODES * HID + 255) / 256, 256, 0, stream>>>(
        agg1, norm, NODES * HID, 7);

    // ---- layer 2: meanpre/logstdpre = (hidden @ {Wm,Ws}) * norm ; scatter ; reparam ----
    split_f16<<<(NODES * HID + 255) / 256, 256, 0, stream>>>(agg1, Hh, Hl, NODES * HID);
    split_f16<<<(HID * DZ + 255) / 256, 256, 0, stream>>>(Wm, Wmh, Wml, HID * DZ);
    split_f16<<<(HID * DZ + 255) / 256, 256, 0, stream>>>(Wsd, Wsh, Wsl, HID * DZ);
    gemm_wmma_split<false, true, false><<<dim3(1, NODES / 16), blk, 0, stream>>>(
        Hh, Hl, Wmh, Wml, t2, NODES, DZ, HID, norm);
    gemm_wmma_split<false, true, false><<<dim3(1, NODES / 16), blk, 0, stream>>>(
        Hh, Hl, Wsh, Wsl, t2 + (size_t)NODES * DZ, NODES, DZ, HID, norm);
    {
        long long tot = (long long)E * DZ;
        scatter_add<<<(unsigned)((tot + 255) / 256), 256, 0, stream>>>(
            t2, agg2, src, dst, tot, DZ, 6);
        scatter_add<<<(unsigned)((tot + 255) / 256), 256, 0, stream>>>(
            t2 + (size_t)NODES * DZ, agg2 + (size_t)NODES * DZ, src, dst, tot, DZ, 6);
    }
    z_kernel<<<(NODES * DZ + 255) / 256, 256, 0, stream>>>(
        agg2, agg2 + (size_t)NODES * DZ, norm, noise, Zf, NODES * DZ);

    // ---- adj_logits = Z @ Z^T (Z f16 pair ~3MB -> L2 resident; NT-stream 604MB out) ----
    split_f16<<<(NODES * DZ + 255) / 256, 256, 0, stream>>>(Zf, Zh, Zl, NODES * DZ);
    gemm_wmma_split<true, false, true><<<dim3(NODES / 16 / 8, NODES / 16), blk, 0, stream>>>(
        Zh, Zl, Zh, Zl, out, NODES, NODES, DZ, nullptr);
}
